// LSTMModel_49383533969520
// MI455X (gfx1250) — matrix operand order repaired and checked by reading, not compiled, not run on hardware
//
#include <hip/hip_runtime.h>
#include <hip/hip_bf16.h>

// ---------------------------------------------------------------------------
// 2-layer LSTM, B=64, T=512, D=H=512, f32 in/out, bf16 WMMA internally.
// ---------------------------------------------------------------------------
#define BATCH 64
#define TSTEPS 512
#define DIN 512
#define HID 512
#define G4 2048                  // 4*H gate width
#define MROWS (TSTEPS * BATCH)   // 32768 rows of the big input-projection GEMM

#define NWG_RECUR 16       // persistent workgroups for recurrence
#define WH_STRIDE 520      // 512 + 8 bf16 pad (breaks LDS bank alignment)
#define H_STRIDE  520      // padded LDS stride for staged h_prev
#define GL_STRIDE 132      // 128 + 4 f32 pad for gate staging buffer
#define SMEM_WH   (128 * WH_STRIDE * 2)
#define SMEM_H    (BATCH * H_STRIDE * 2)
#define SMEM_G    (BATCH * GL_STRIDE * 4)
#define SMEM_C    (BATCH * 32 * 4)
#define SMEM_RECUR (SMEM_WH + SMEM_H + SMEM_G + SMEM_C)   // 241,664 B < 320 KB

typedef __bf16 bf16;
typedef __attribute__((ext_vector_type(16))) __bf16 v16bf;
typedef __attribute__((ext_vector_type(8)))  __bf16 v8bf;
typedef __attribute__((ext_vector_type(8)))  float  v8f;

union V16 { v16bf v; v8bf h[2]; };

// A fragment: 16x32 bf16 tile, row-major, stride in elements.
// Lane l holds row m=l&15; lanes 0-15: K in {0..7}u{16..23}, lanes 16-31: +8.
__device__ __forceinline__ v16bf loadA16x32(const bf16* base, int stride, int lane, int k0) {
  const int m  = lane & 15;
  const int kb = (lane & 16) ? 8 : 0;
  const bf16* p = base + (size_t)m * stride + (k0 + kb);
  V16 u;
  u.h[0] = *(const v8bf*)(p);
  u.h[1] = *(const v8bf*)(p + 16);
  return u.v;
}

// B fragment: 32x16 tile where B column n is a contiguous memory row (W row n).
// Lane l holds col n=l&15; lanes 0-15: K in {0..7}u{16..23}, lanes 16-31: +8.
__device__ __forceinline__ v16bf loadB32x16(const bf16* base, int stride, int lane, int k0) {
  const int n  = lane & 15;
  const int kb = (lane & 16) ? 8 : 0;
  const bf16* p = base + (size_t)n * stride + (k0 + kb);
  V16 u;
  u.h[0] = *(const v8bf*)(p);
  u.h[1] = *(const v8bf*)(p + 16);
  return u.v;
}

__device__ __forceinline__ float sigmoidf_(float x) {
  return 1.0f / (1.0f + __expf(-x));
}

// ---------------------------------------------------------------------------
// f32 -> bf16 flat convert (weights)
// ---------------------------------------------------------------------------
__global__ void cvt_f32_bf16(const float* __restrict__ in, bf16* __restrict__ out, int n) {
  int i = blockIdx.x * blockDim.x + threadIdx.x;
  if (i < n) out[i] = (bf16)in[i];
}

// x: [B,T,D] f32  ->  xb: [T*B, D] bf16 (time-major transpose)
__global__ void cvt_x_tmajor(const float* __restrict__ x, bf16* __restrict__ xb) {
  const int row = blockIdx.x;          // row = t*B + b
  const int t = row >> 6;
  const int b = row & 63;
  const float* src = x + ((size_t)b * TSTEPS + t) * DIN;
  bf16* dst = xb + (size_t)row * DIN;
  for (int d = threadIdx.x; d < DIN; d += blockDim.x)
    dst[d] = (bf16)src[d];
}

// ---------------------------------------------------------------------------
// Big input-projection GEMM:  C[M,2048] = A[M,512] * W^T + (b0 + b1)
// A row-major bf16, W row-major bf16 [2048,512] (cols of B contiguous).
// 8 waves / block; each wave: TWO 16-row m-tiles x 128-col strip
// (16 accumulators; every B fragment reused for 2 wmma -> ~1.6x L2 intensity).
// grid.x = (M/32) * 16 strips / 8 waves = 2048 blocks for M=32768.
// ---------------------------------------------------------------------------
__global__ void __launch_bounds__(256, 1)
gemm_xg(const bf16* __restrict__ A, const bf16* __restrict__ W,
        const float* __restrict__ bias0, const float* __restrict__ bias1,
        float* __restrict__ C) {
  const int tid  = threadIdx.x;
  const int wave = tid >> 5;
  const int lane = tid & 31;
  const int wid  = blockIdx.x * 8 + wave;
  const int mt    = (wid >> 4) * 2;  // first of two 16-row tiles
  const int strip = wid & 15;        // 0..15    (128-col strip)
  const int n0 = strip * 128;

  const bf16* Arow0 = A + (size_t)mt * 16 * DIN;
  const bf16* Arow1 = Arow0 + (size_t)16 * DIN;

  v8f acc0[8] = {};
  v8f acc1[8] = {};
  for (int k0 = 0; k0 < DIN; k0 += 32) {
    // prefetch streaming A operand ~4 k-steps ahead (global_prefetch_b8)
    __builtin_prefetch((const void*)(Arow0 + (size_t)(lane & 15) * DIN + k0 + 128), 0, 1);
    __builtin_prefetch((const void*)(Arow1 + (size_t)(lane & 15) * DIN + k0 + 128), 0, 1);
    v16bf a0 = loadA16x32(Arow0, DIN, lane, k0);
    v16bf a1 = loadA16x32(Arow1, DIN, lane, k0);
#pragma unroll
    for (int j = 0; j < 8; ++j) {
      v16bf b = loadB32x16(W + (size_t)(n0 + j * 16) * DIN, DIN, lane, k0);
      acc0[j] = __builtin_amdgcn_wmma_f32_16x16x32_bf16(
          false, a0, false, b, (short)0, acc0[j], false, false);
      acc1[j] = __builtin_amdgcn_wmma_f32_16x16x32_bf16(
          false, a1, false, b, (short)0, acc1[j], false, false);
    }
  }

  const int mbase0 = mt * 16 + ((lane & 16) ? 8 : 0);
  const int mbase1 = mbase0 + 16;
#pragma unroll
  for (int j = 0; j < 8; ++j) {
    const int col = n0 + j * 16 + (lane & 15);
    const float bia = bias0[col] + bias1[col];
#pragma unroll
    for (int r = 0; r < 8; ++r) {
      C[(size_t)(mbase0 + r) * G4 + col] = acc0[j][r] + bia;
      C[(size_t)(mbase1 + r) * G4 + col] = acc1[j][r] + bia;
    }
  }
}

// ---------------------------------------------------------------------------
// Device-wide barrier for the persistent recurrence grid (16 WGs).
// Counter is zeroed by hipMemsetAsync before launch; target grows monotonically.
// ---------------------------------------------------------------------------
__device__ __forceinline__ void gridBarrier(unsigned* cnt, unsigned target) {
  __threadfence();
  __syncthreads();
  if (threadIdx.x == 0) {
    __hip_atomic_fetch_add(cnt, 1u, __ATOMIC_RELEASE, __HIP_MEMORY_SCOPE_AGENT);
    while (__hip_atomic_load(cnt, __ATOMIC_ACQUIRE, __HIP_MEMORY_SCOPE_AGENT) < target)
      __builtin_amdgcn_s_sleep(1);
  }
  __syncthreads();
}

// ---------------------------------------------------------------------------
// Persistent LSTM recurrence. 16 WGs x 256 threads. WG `wg` owns h columns
// [wg*32, wg*32+32). Its 128 Wh rows (f/i/g/o blocks) stay in LDS for all T
// steps; h_prev is staged into LDS once per step (8x less L2 traffic than
// per-wave global reads); c-state stays in LDS; h ping-pongs through L2 with
// one grid barrier per timestep.
//   gates[64,128] = h_prev[64,512] @ WhSlice^T + xg[t] slice
// ---------------------------------------------------------------------------
__global__ void __launch_bounds__(256, 1)
lstm_recur(const bf16* __restrict__ Whb,      // [2048,512] bf16
           const float* __restrict__ xg,      // [T*B, 2048] f32 (includes biases)
           bf16* __restrict__ hping,          // [64,512] bf16 (zeroed)
           bf16* __restrict__ hpong,          // [64,512] bf16 (zeroed)
           bf16* __restrict__ hseq,           // [T*B,512] bf16 or nullptr
           float* __restrict__ hfinal,        // [64,512] f32 or nullptr
           unsigned* __restrict__ sync) {
  extern __shared__ char smem[];
  bf16*  whLds = (bf16*)smem;                          // 128 x 520 bf16
  bf16*  hLds  = (bf16*)(smem + SMEM_WH);              // 64 x 520 bf16
  float* gLds  = (float*)(smem + SMEM_WH + SMEM_H);    // 64 x 132 f32
  float* cLds  = gLds + BATCH * GL_STRIDE;             // 64 x 32 f32

  const int wg   = blockIdx.x;     // 0..15
  const int tid  = threadIdx.x;
  const int wave = tid >> 5;
  const int lane = tid & 31;
  const int j0   = wg * 32;        // first owned h-column

  // --- load this WG's 128 Wh rows into LDS (row r -> gate block r>>5, col j0+(r&31))
  for (int r = wave; r < 128; r += 8) {
    const int grow = (r >> 5) * HID + j0 + (r & 31);
    const bf16* src = Whb + (size_t)grow * HID;
    bf16* dst = whLds + (size_t)r * WH_STRIDE;
    const v8bf* s8 = (const v8bf*)src;
    v8bf* d8 = (v8bf*)dst;
    d8[lane * 2]     = s8[lane * 2];
    d8[lane * 2 + 1] = s8[lane * 2 + 1];
  }
  // --- zero c state
  for (int i = tid; i < BATCH * 32; i += 256) cLds[i] = 0.0f;
  __syncthreads();

  // wave -> local gate-column tile [16*wave, 16*wave+16)
  const int nloc   = wave * 16 + (lane & 15);
  const int gcol   = (nloc >> 5) * HID + j0 + (nloc & 31); // global xg column
  const int mbase  = (lane & 16) ? 8 : 0;
  const bf16* wB   = whLds + (size_t)(wave * 16) * WH_STRIDE;

  for (int t = 0; t < TSTEPS; ++t) {
    const bf16* hprev = (t & 1) ? hpong : hping;
    bf16*       hnext = (t & 1) ? hping : hpong;

    // ---- stage h_prev [64,512] into LDS (cooperative, 16B chunks)
    for (int idx = tid; idx < BATCH * (HID / 8); idx += 256) {
      const int r = idx >> 6;             // row 0..63
      const int c = (idx & 63) * 8;       // col chunk
      *(v8bf*)(hLds + (size_t)r * H_STRIDE + c) =
          *(const v8bf*)(hprev + (size_t)r * HID + c);
    }
    __syncthreads();

    // ---- gates GEMM: [64,512] x [512,16] per wave, 4 m-tiles, A from LDS
    v8f acc[4] = {};
    for (int k0 = 0; k0 < HID; k0 += 32) {
      v16bf b = loadB32x16(wB, WH_STRIDE, lane, k0);
#pragma unroll
      for (int m4 = 0; m4 < 4; ++m4) {
        v16bf a = loadA16x32(hLds + (size_t)m4 * 16 * H_STRIDE, H_STRIDE, lane, k0);
        acc[m4] = __builtin_amdgcn_wmma_f32_16x16x32_bf16(
            false, a, false, b, (short)0, acc[m4], false, false);
      }
    }

    // ---- add precomputed input gates, stage to LDS
    const float* xgt = xg + (size_t)t * BATCH * G4;
#pragma unroll
    for (int m4 = 0; m4 < 4; ++m4) {
#pragma unroll
      for (int r = 0; r < 8; ++r) {
        const int b = m4 * 16 + mbase + r;
        gLds[b * GL_STRIDE + nloc] = acc[m4][r] + xgt[(size_t)b * G4 + gcol];
      }
    }
    __syncthreads();

    // ---- elementwise LSTM cell update for owned 64x32 slice
    for (int i = tid; i < BATCH * 32; i += 256) {
      const int b = i >> 5;
      const int j = i & 31;
      const float fg = sigmoidf_(gLds[b * GL_STRIDE + j]);
      const float ig = sigmoidf_(gLds[b * GL_STRIDE + 32 + j]);
      const float gg = tanhf(gLds[b * GL_STRIDE + 64 + j]);
      const float og = sigmoidf_(gLds[b * GL_STRIDE + 96 + j]);
      const float c  = fg * cLds[i] + ig * gg;
      cLds[i] = c;
      const float h = og * tanhf(c);
      hnext[(size_t)b * HID + j0 + j] = (bf16)h;
      if (hseq)   hseq[((size_t)t * BATCH + b) * HID + j0 + j] = (bf16)h;
      if (hfinal && t == TSTEPS - 1) hfinal[(size_t)b * HID + j0 + j] = h;
    }

    gridBarrier(sync, (unsigned)(NWG_RECUR * (t + 1)));
  }
}

// ---------------------------------------------------------------------------
// Host launcher
// ---------------------------------------------------------------------------
extern "C" void kernel_launch(void* const* d_in, const int* in_sizes, int n_in,
                              void* d_out, int out_size, void* d_ws, size_t ws_size,
                              hipStream_t stream) {
  const float* x   = (const float*)d_in[0];
  const float* Wx0 = (const float*)d_in[1];
  const float* bx0 = (const float*)d_in[2];
  const float* Wh0 = (const float*)d_in[3];
  const float* bh0 = (const float*)d_in[4];
  const float* Wx1 = (const float*)d_in[5];
  const float* bx1 = (const float*)d_in[6];
  const float* Wh1 = (const float*)d_in[7];
  const float* bh1 = (const float*)d_in[8];
  float* out = (float*)d_out;

  // workspace carve-up (256B aligned)
  char* ws = (char*)d_ws;
  size_t off = 0;
  auto take = [&](size_t bytes) -> char* {
    char* p = ws + off;
    off += (bytes + 255) & ~(size_t)255;
    return p;
  };
  bf16*  xb    = (bf16*)take((size_t)MROWS * DIN * 2);      // 33.5 MB
  bf16*  hseq  = (bf16*)take((size_t)MROWS * HID * 2);      // 33.5 MB
  bf16*  Wxb0  = (bf16*)take((size_t)G4 * DIN * 2);
  bf16*  Whb0  = (bf16*)take((size_t)G4 * HID * 2);
  bf16*  Wxb1  = (bf16*)take((size_t)G4 * HID * 2);
  bf16*  Whb1  = (bf16*)take((size_t)G4 * HID * 2);
  float* xgbuf = (float*)take((size_t)MROWS * G4 * 4);      // 268 MB (reused L0/L1)
  bf16*  hping = (bf16*)take((size_t)BATCH * HID * 2);
  bf16*  hpong = (bf16*)take((size_t)BATCH * HID * 2);
  unsigned* syncc = (unsigned*)take(256);

  (void)in_sizes; (void)n_in; (void)out_size; (void)ws_size;

  // allow >64KB dynamic LDS for the persistent recurrence kernel
  hipFuncSetAttribute((const void*)lstm_recur,
                      hipFuncAttributeMaxDynamicSharedMemorySize, SMEM_RECUR);

  const int WN = G4 * DIN; // weight element count

  // 1) converts
  cvt_x_tmajor<<<MROWS, 256, 0, stream>>>(x, xb);
  cvt_f32_bf16<<<(WN + 255) / 256, 256, 0, stream>>>(Wx0, Wxb0, WN);
  cvt_f32_bf16<<<(WN + 255) / 256, 256, 0, stream>>>(Wh0, Whb0, WN);
  cvt_f32_bf16<<<(WN + 255) / 256, 256, 0, stream>>>(Wx1, Wxb1, WN);
  cvt_f32_bf16<<<(WN + 255) / 256, 256, 0, stream>>>(Wh1, Whb1, WN);

  // 2) layer-0 input projection: xg = xb @ Wx0^T + bx0 + bh0
  gemm_xg<<<(MROWS / 32) * 16 / 8, 256, 0, stream>>>(xb, Wxb0, bx0, bh0, xgbuf);

  // 3) layer-0 recurrence (persistent, 16 WGs)
  hipMemsetAsync(hping, 0, (size_t)BATCH * HID * 2, stream);
  hipMemsetAsync(hpong, 0, (size_t)BATCH * HID * 2, stream);
  hipMemsetAsync(syncc, 0, 256, stream);
  lstm_recur<<<NWG_RECUR, 256, SMEM_RECUR, stream>>>(
      Whb0, xgbuf, hping, hpong, hseq, nullptr, syncc);

  // 4) layer-1 input projection: xg = hseq @ Wx1^T + bx1 + bh1
  gemm_xg<<<(MROWS / 32) * 16 / 8, 256, 0, stream>>>(hseq, Wxb1, bx1, bh1, xgbuf);

  // 5) layer-1 recurrence; final h -> d_out (f32 [64,512])
  hipMemsetAsync(hping, 0, (size_t)BATCH * HID * 2, stream);
  hipMemsetAsync(hpong, 0, (size_t)BATCH * HID * 2, stream);
  hipMemsetAsync(syncc, 0, 256, stream);
  lstm_recur<<<NWG_RECUR, 256, SMEM_RECUR, stream>>>(
      Whb1, xgbuf, hping, hpong, nullptr, out, syncc);
}
